// StabilizedLayer_82300163326553
// MI455X (gfx1250) — matrix operand order repaired and checked by reading, not compiled, not run on hardware
//
#include <hip/hip_runtime.h>

// ---------------------------------------------------------------------------
// Problem constants (from reference: B=2, L=2048, D=2048, H=16, F=4D)
// ---------------------------------------------------------------------------
constexpr int Bsz = 2;
constexpr int L   = 2048;
constexpr int D   = 2048;
constexpr int H   = 16;
constexpr int DH  = 128;        // D / H
constexpr int F   = 4 * D;      // 8192
constexpr int TD  = 3 * D;      // 6144
constexpr int Mrows = Bsz * L;  // 4096

typedef __bf16 bf16;
typedef __attribute__((ext_vector_type(16))) __bf16 v16bf;
typedef __attribute__((ext_vector_type(8)))  __bf16 v8bf;
typedef __attribute__((ext_vector_type(4)))  __bf16 v4bf;
typedef __attribute__((ext_vector_type(8)))  float  v8f;
typedef __attribute__((__vector_size__(16))) int    v4i;

typedef __attribute__((address_space(1))) v4i* gv4ip;  // global int4*
typedef __attribute__((address_space(3))) v4i* lv4ip;  // LDS int4*

// ---------------------------------------------------------------------------
// CDNA5 async global->LDS copy (ASYNCcnt path), with sync fallback.
// Builtin signature (probe-confirmed by diagnostic): (v4i as1* src,
// v4i as3* dst, imm offset, imm cpol).
// ---------------------------------------------------------------------------
#if defined(__has_builtin)
#if __has_builtin(__builtin_amdgcn_global_load_async_to_lds_b128)
#define HAS_ASYNC_LDS 1
#endif
#endif
#ifndef HAS_ASYNC_LDS
#define HAS_ASYNC_LDS 0
#endif

__device__ __forceinline__ void copy16_g2l(bf16* dst, const bf16* src) {
#if HAS_ASYNC_LDS
  __builtin_amdgcn_global_load_async_to_lds_b128((gv4ip)src, (lv4ip)dst, 0, 0);
#else
  *(v8bf*)dst = *(const v8bf*)src;
#endif
}

template <int N>
__device__ __forceinline__ void wait_async() {
#if HAS_ASYNC_LDS
  asm volatile("s_wait_asynccnt %0" ::"i"(N) : "memory");
#endif
}

__device__ __forceinline__ float fast_rcp(float x) {
#if defined(__has_builtin)
#if __has_builtin(__builtin_amdgcn_rcpf)
  return __builtin_amdgcn_rcpf(x);
#else
  return 1.0f / x;
#endif
#else
  return 1.0f / x;
#endif
}

// ---------------------------------------------------------------------------
// WMMA helpers. CDNA5 v_wmma_f32_16x16x32_bf16:  D = A(16x32) * B(32x16) + C
// A layout (16-bit, ISA 7.12.2): lane m=l&15, g=l>>4; VGPR0-3 hold K=8g..8g+7,
// VGPR4-7 hold K=16+8g..23+8g  -> two contiguous 8-half (b128) reads.
// B layout (16-bit): lane n=l&15, g=l>>4; same K order as A: K=8g..8g+7 and
// K=16+8g..23+8g -> two contiguous 8-half reads.
// C/D layout: lane n=l&15, VGPR r holds row m = r + 8*(l>>4).
// ---------------------------------------------------------------------------
__device__ __forceinline__ v8f wmma_bf16(v16bf a, v16bf b, v8f c) {
  return __builtin_amdgcn_wmma_f32_16x16x32_bf16(
      false, a, false, b, (short)0, c, false, false);
}

__device__ __forceinline__ v16bf ldfragA(const bf16* p, int g) {
  v8bf lo = *(const v8bf*)(p + 8 * g);
  v8bf hi = *(const v8bf*)(p + 16 + 8 * g);
  return __builtin_shufflevector(lo, hi, 0, 1, 2, 3, 4, 5, 6, 7,
                                 8, 9, 10, 11, 12, 13, 14, 15);
}

__device__ __forceinline__ v16bf ldfragB(const bf16* p, int g) {
  v8bf lo = *(const v8bf*)(p + 8 * g);
  v8bf hi = *(const v8bf*)(p + 16 + 8 * g);
  return __builtin_shufflevector(lo, hi, 0, 1, 2, 3, 4, 5, 6, 7,
                                 8, 9, 10, 11, 12, 13, 14, 15);
}

// ---------------------------------------------------------------------------
// fp32 -> bf16 conversion (weights), vectorized
// ---------------------------------------------------------------------------
__global__ __launch_bounds__(256) void cvt_f32_bf16(const float* __restrict__ in,
                                                    bf16* __restrict__ out, int n4) {
  for (int i = blockIdx.x * blockDim.x + threadIdx.x; i < n4;
       i += gridDim.x * blockDim.x) {
    float4 v = ((const float4*)in)[i];
    v4bf o;
    o.x = (bf16)v.x; o.y = (bf16)v.y; o.z = (bf16)v.z; o.w = (bf16)v.w;
    ((v4bf*)out)[i] = o;
  }
}

// ---------------------------------------------------------------------------
// LayerNorm: one 256-thread block per row of D=2048, fp32 in -> bf16 out
// ---------------------------------------------------------------------------
__global__ __launch_bounds__(256) void ln_bf16(const float* __restrict__ x,
                                               const float* __restrict__ gw,
                                               const float* __restrict__ bw,
                                               bf16* __restrict__ out) {
  const int row = blockIdx.x;
  const int t = threadIdx.x;
  const float* xr = x + (size_t)row * D;
  float vals[8];
  float s1 = 0.f, s2 = 0.f;
#pragma unroll
  for (int i = 0; i < 8; ++i) {
    float v = xr[t + i * 256];
    vals[i] = v;
    s1 += v;
    s2 += v * v;
  }
#pragma unroll
  for (int off = 1; off < 32; off <<= 1) {
    s1 += __shfl_xor(s1, off, 32);
    s2 += __shfl_xor(s2, off, 32);
  }
  __shared__ float r1[8], r2[8];
  int wid = t >> 5, lane = t & 31;
  if (lane == 0) { r1[wid] = s1; r2[wid] = s2; }
  __syncthreads();
  float t1 = 0.f, t2 = 0.f;
#pragma unroll
  for (int i = 0; i < 8; ++i) { t1 += r1[i]; t2 += r2[i]; }
  const float mean = t1 * (1.0f / D);
  const float var  = t2 * (1.0f / D) - mean * mean;
  const float rs   = rsqrtf(var + 1e-6f);
  bf16* orow = out + (size_t)row * D;
#pragma unroll
  for (int i = 0; i < 8; ++i) {
    int c = t + i * 256;
    orow[c] = (bf16)((vals[i] - mean) * rs * gw[c] + bw[c]);
  }
}

// ---------------------------------------------------------------------------
// Transpose V out of the packed qkv buffer: vT[b][h][dh][l] = qkv[b][l][2D+h*Dh+dh]
// ---------------------------------------------------------------------------
__global__ __launch_bounds__(256) void transpose_v(const bf16* __restrict__ qkv,
                                                   bf16* __restrict__ vT) {
  const int total = Bsz * L * D;
  for (int idx = blockIdx.x * blockDim.x + threadIdx.x; idx < total;
       idx += gridDim.x * blockDim.x) {
    int b = idx / (L * D);
    int rem = idx - b * (L * D);
    int l = rem / D;
    int d = rem - l * D;
    int h = d / DH;
    int dh = d - h * DH;
    bf16 v = qkv[(size_t)(b * L + l) * TD + 2 * D + d];
    vT[((size_t)(b * H + h) * DH + dh) * L + l] = v;
  }
}

// ---------------------------------------------------------------------------
// Tiled WMMA GEMM:  OUT[M,N] = A[M,K] * W[N,K]^T   (both bf16, fp32 accumulate)
// Block tile 128x128, K-step 32, 8 waves (2M x 4N), wave tile 64x32.
// Double-buffered LDS; tiles arrive via async global->LDS DMA (ASYNCcnt).
// EPI 0: store bf16 into outb.   EPI 1: outf = resid + acc (fp32).
// ---------------------------------------------------------------------------
template <int EPI>
__global__ __launch_bounds__(256) void gemm_wmma(const bf16* __restrict__ A,
                                                 const bf16* __restrict__ W,
                                                 const float* __restrict__ resid,
                                                 float* __restrict__ outf,
                                                 bf16* __restrict__ outb,
                                                 int M, int N, int K) {
  __shared__ bf16 sA[2][128 * 32];
  __shared__ bf16 sB[2][128 * 32];
  const int tid  = threadIdx.x;
  const int wid  = tid >> 5;
  const int lane = tid & 31;
  const int n15  = lane & 15;
  const int g    = lane >> 4;
  const int wm   = wid & 1;   // 0..1 -> 64-row half
  const int wn   = wid >> 1;  // 0..3 -> 32-col slice
  const int m0   = blockIdx.y * 128;
  const int n0   = blockIdx.x * 128;

  // per-tile-pair: each thread issues 2 A + 2 B copies (4 async ops/wave-slot)
  auto load_tiles = [&](int kt, int buf) {
#pragma unroll
    for (int it = 0; it < 2; ++it) {
      int i = tid + it * 256;
      int r_ = i >> 2, c = i & 3;
      copy16_g2l(&sA[buf][r_ * 32 + c * 8],
                 A + (size_t)(m0 + r_) * K + kt * 32 + c * 8);
      copy16_g2l(&sB[buf][r_ * 32 + c * 8],
                 W + (size_t)(n0 + r_) * K + kt * 32 + c * 8);
    }
  };

  v8f acc[4][2];
  const v8f vz = {0.f, 0.f, 0.f, 0.f, 0.f, 0.f, 0.f, 0.f};
#pragma unroll
  for (int i = 0; i < 4; ++i)
#pragma unroll
    for (int j = 0; j < 2; ++j) acc[i][j] = vz;

  const int nk = K >> 5;
  load_tiles(0, 0);
  for (int kt = 0; kt < nk; ++kt) {
    const int cur = kt & 1;
    if (kt + 1 < nk) {
      load_tiles(kt + 1, cur ^ 1);
      wait_async<4>();   // retire the older (current-tile) group only
    } else {
      wait_async<0>();
    }
    __syncthreads();

    v16bf a[4];
#pragma unroll
    for (int ms = 0; ms < 4; ++ms)
      a[ms] = ldfragA(&sA[cur][(wm * 64 + ms * 16 + n15) * 32], g);
#pragma unroll
    for (int ns = 0; ns < 2; ++ns) {
      v16bf b = ldfragB(&sB[cur][(wn * 32 + ns * 16 + n15) * 32], g);
#pragma unroll
      for (int ms = 0; ms < 4; ++ms) acc[ms][ns] = wmma_bf16(a[ms], b, acc[ms][ns]);
    }
    __syncthreads();
  }

#pragma unroll
  for (int ms = 0; ms < 4; ++ms)
#pragma unroll
    for (int ns = 0; ns < 2; ++ns)
#pragma unroll
      for (int r = 0; r < 8; ++r) {
        int row = m0 + wm * 64 + ms * 16 + 8 * g + r;
        int col = n0 + wn * 32 + ns * 16 + n15;
        size_t idx = (size_t)row * N + col;
        if (EPI == 0) outb[idx] = (bf16)acc[ms][ns][r];
        else          outf[idx] = resid[idx] + acc[ms][ns][r];
      }
}

// ---------------------------------------------------------------------------
// Fused gate/up GEMM + SiLU(gate)*up -> bf16 (double-buffered, async tiles)
// ---------------------------------------------------------------------------
__global__ __launch_bounds__(256) void gemm_gateup(const bf16* __restrict__ A,
                                                   const bf16* __restrict__ Wg,
                                                   const bf16* __restrict__ Wu,
                                                   bf16* __restrict__ outb,
                                                   int M, int N, int K) {
  __shared__ bf16 sA[2][128 * 32];
  __shared__ bf16 sG[2][128 * 32];
  __shared__ bf16 sU[2][128 * 32];
  const int tid  = threadIdx.x;
  const int wid  = tid >> 5;
  const int lane = tid & 31;
  const int n15  = lane & 15;
  const int g    = lane >> 4;
  const int wm   = wid & 1;
  const int wn   = wid >> 1;
  const int m0   = blockIdx.y * 128;
  const int n0   = blockIdx.x * 128;

  auto load_tiles = [&](int kt, int buf) {
#pragma unroll
    for (int it = 0; it < 2; ++it) {
      int i = tid + it * 256;
      int r_ = i >> 2, c = i & 3;
      copy16_g2l(&sA[buf][r_ * 32 + c * 8],
                 A + (size_t)(m0 + r_) * K + kt * 32 + c * 8);
      copy16_g2l(&sG[buf][r_ * 32 + c * 8],
                 Wg + (size_t)(n0 + r_) * K + kt * 32 + c * 8);
      copy16_g2l(&sU[buf][r_ * 32 + c * 8],
                 Wu + (size_t)(n0 + r_) * K + kt * 32 + c * 8);
    }
  };

  v8f ag[4][2], au[4][2];
  const v8f vz = {0.f, 0.f, 0.f, 0.f, 0.f, 0.f, 0.f, 0.f};
#pragma unroll
  for (int i = 0; i < 4; ++i)
#pragma unroll
    for (int j = 0; j < 2; ++j) { ag[i][j] = vz; au[i][j] = vz; }

  const int nk = K >> 5;
  load_tiles(0, 0);
  for (int kt = 0; kt < nk; ++kt) {
    const int cur = kt & 1;
    if (kt + 1 < nk) {
      load_tiles(kt + 1, cur ^ 1);
      wait_async<6>();   // 6 async ops per tile-triple group
    } else {
      wait_async<0>();
    }
    __syncthreads();

    v16bf a[4];
#pragma unroll
    for (int ms = 0; ms < 4; ++ms)
      a[ms] = ldfragA(&sA[cur][(wm * 64 + ms * 16 + n15) * 32], g);
#pragma unroll
    for (int ns = 0; ns < 2; ++ns) {
      v16bf bg = ldfragB(&sG[cur][(wn * 32 + ns * 16 + n15) * 32], g);
      v16bf bu = ldfragB(&sU[cur][(wn * 32 + ns * 16 + n15) * 32], g);
#pragma unroll
      for (int ms = 0; ms < 4; ++ms) {
        ag[ms][ns] = wmma_bf16(a[ms], bg, ag[ms][ns]);
        au[ms][ns] = wmma_bf16(a[ms], bu, au[ms][ns]);
      }
    }
    __syncthreads();
  }

#pragma unroll
  for (int ms = 0; ms < 4; ++ms)
#pragma unroll
    for (int ns = 0; ns < 2; ++ns)
#pragma unroll
      for (int r = 0; r < 8; ++r) {
        int row = m0 + wm * 64 + ms * 16 + 8 * g + r;
        int col = n0 + wn * 32 + ns * 16 + n15;
        float gv = ag[ms][ns][r];
        float uv = au[ms][ns][r];
        float silu = gv * fast_rcp(1.0f + __expf(-gv));
        outb[(size_t)row * N + col] = (bf16)(silu * uv);
      }
}

// ---------------------------------------------------------------------------
// Flash attention: one wave per (b, h, 16-query block). 32-key blocks.
// Q,K fragments straight from the packed qkv buffer (contiguous along d);
// V fragments from vT (contiguous along l). P round-trips through per-wave LDS.
// ---------------------------------------------------------------------------
__global__ __launch_bounds__(128) void flash_attn(const bf16* __restrict__ qkv,
                                                  const bf16* __restrict__ vT,
                                                  bf16* __restrict__ attnout) {
  __shared__ bf16 sP[4][16 * 32];
  const int wid  = threadIdx.x >> 5;
  const int lane = threadIdx.x & 31;
  const int n15  = lane & 15;
  const int g    = lane >> 4;
  const int wgid = blockIdx.x * 4 + wid;   // 0 .. 4095
  const int qb   = wgid & 127;             // query block (L/16)
  const int bh   = wgid >> 7;              // b*H + h
  const int h    = bh & (H - 1);
  const int b    = bh >> 4;

  // Q fragments: row m = n15 of this q-block
  const bf16* qp = qkv + (size_t)(b * L + qb * 16 + n15) * TD + h * DH;
  v16bf aq[4];
#pragma unroll
  for (int c = 0; c < 4; ++c) aq[c] = ldfragA(qp + 32 * c, g);

  v8f o[8];
  const v8f vz = {0.f, 0.f, 0.f, 0.f, 0.f, 0.f, 0.f, 0.f};
#pragma unroll
  for (int i = 0; i < 8; ++i) o[i] = vz;
  float mrow[8], lsum[8];
#pragma unroll
  for (int r = 0; r < 8; ++r) { mrow[r] = -1e30f; lsum[r] = 0.f; }

  const float scale = 0.08838834764831845f;  // 1/sqrt(128)
  const int kbmax = (qb * 16 + 15) >> 5;

  for (int kb = 0; kb <= kbmax; ++kb) {
    // S = Q * K^T  (two 16-col groups of a 32-key block)
    v8f s[2];
#pragma unroll
    for (int t = 0; t < 2; ++t) {
      s[t] = vz;
      const bf16* kp =
          qkv + (size_t)(b * L + kb * 32 + t * 16 + n15) * TD + D + h * DH;
#pragma unroll
      for (int c = 0; c < 4; ++c) {
        v16bf bk = ldfragB(kp + 32 * c, g);  // B = K^T, K-dim = d (contiguous)
        s[t] = wmma_bf16(aq[c], bk, s[t]);
      }
    }

    // scale + causal mask + online softmax
    float mnew[8], fac[8];
#pragma unroll
    for (int r = 0; r < 8; ++r) {
      const int qi = qb * 16 + 8 * g + r;
#pragma unroll
      for (int t = 0; t < 2; ++t) {
        int ki = kb * 32 + t * 16 + n15;
        float v = s[t][r] * scale;
        if (ki > qi) v = -1e30f;
        s[t][r] = v;
      }
      float v = fmaxf(s[0][r], s[1][r]);
#pragma unroll
      for (int off = 1; off < 16; off <<= 1) v = fmaxf(v, __shfl_xor(v, off, 32));
      mnew[r] = fmaxf(mrow[r], v);
      fac[r]  = __expf(mrow[r] - mnew[r]);
    }
#pragma unroll
    for (int r = 0; r < 8; ++r) {
      float ps = 0.f;
#pragma unroll
      for (int t = 0; t < 2; ++t) {
        float p = __expf(s[t][r] - mnew[r]);
        s[t][r] = p;
        ps += p;
      }
#pragma unroll
      for (int off = 1; off < 16; off <<= 1) ps += __shfl_xor(ps, off, 32);
      lsum[r] = lsum[r] * fac[r] + ps;
      mrow[r] = mnew[r];
    }
#pragma unroll
    for (int d8 = 0; d8 < 8; ++d8)
#pragma unroll
      for (int r = 0; r < 8; ++r) o[d8][r] *= fac[r];

    // P: C-layout scatter into LDS, read back as A-layout fragment
    bf16* P = sP[wid];
#pragma unroll
    for (int r = 0; r < 8; ++r)
#pragma unroll
      for (int t = 0; t < 2; ++t)
        P[(8 * g + r) * 32 + t * 16 + n15] = (bf16)s[t][r];
    asm volatile("s_wait_dscnt 0" ::: "memory");
    v16bf pa = ldfragA(P + n15 * 32, g);

    // O += P * V   (V^T rows contiguous along key index)
#pragma unroll
    for (int d8 = 0; d8 < 8; ++d8) {
      const bf16* vp =
          vT + ((size_t)bh * DH + d8 * 16 + n15) * L + kb * 32;
      v16bf bv = ldfragB(vp, g);
      o[d8] = wmma_bf16(pa, bv, o[d8]);
    }
  }

  float inv[8];
#pragma unroll
  for (int r = 0; r < 8; ++r) inv[r] = (lsum[r] > 0.f) ? fast_rcp(lsum[r]) : 0.f;
#pragma unroll
  for (int d8 = 0; d8 < 8; ++d8)
#pragma unroll
    for (int r = 0; r < 8; ++r) {
      size_t row = (size_t)(b * L + qb * 16 + 8 * g + r);
      attnout[row * D + h * DH + d8 * 16 + n15] = (bf16)(o[d8][r] * inv[r]);
    }
}

// ---------------------------------------------------------------------------
// Host orchestration
// ---------------------------------------------------------------------------
extern "C" void kernel_launch(void* const* d_in, const int* in_sizes, int n_in,
                              void* d_out, int out_size, void* d_ws, size_t ws_size,
                              hipStream_t stream) {
  const float* x      = (const float*)d_in[0];
  const float* W_qkv  = (const float*)d_in[1];
  const float* W_o    = (const float*)d_in[2];
  const float* W_gate = (const float*)d_in[3];
  const float* W_up   = (const float*)d_in[4];
  const float* W_down = (const float*)d_in[5];
  const float* ln1_g  = (const float*)d_in[6];
  const float* ln1_b  = (const float*)d_in[7];
  const float* ln2_g  = (const float*)d_in[8];
  const float* ln2_b  = (const float*)d_in[9];
  float* out = (float*)d_out;

  char* p = (char*)d_ws;
  auto alloc = [&](size_t bytes) -> void* {
    void* r = (void*)p;
    p += (bytes + 255) & ~(size_t)255;
    return r;
  };
  bf16* wq  = (bf16*)alloc((size_t)TD * D * 2);
  bf16* wo  = (bf16*)alloc((size_t)D * D * 2);
  bf16* wg  = (bf16*)alloc((size_t)F * D * 2);
  bf16* wu  = (bf16*)alloc((size_t)F * D * 2);
  bf16* wd  = (bf16*)alloc((size_t)D * F * 2);
  bf16* hbf = (bf16*)alloc((size_t)Mrows * D * 2);
  bf16* qkv = (bf16*)alloc((size_t)Mrows * TD * 2);
  bf16* vT  = (bf16*)alloc((size_t)Mrows * D * 2);
  bf16* att = (bf16*)alloc((size_t)Mrows * D * 2);
  float* x2 = (float*)alloc((size_t)Mrows * D * 4);
  bf16* gu  = (bf16*)alloc((size_t)Mrows * F * 2);

  // 1. weights -> bf16
  cvt_f32_bf16<<<2048, 256, 0, stream>>>(W_qkv,  wq, TD * D / 4);
  cvt_f32_bf16<<<2048, 256, 0, stream>>>(W_o,    wo, D * D / 4);
  cvt_f32_bf16<<<2048, 256, 0, stream>>>(W_gate, wg, F * D / 4);
  cvt_f32_bf16<<<2048, 256, 0, stream>>>(W_up,   wu, F * D / 4);
  cvt_f32_bf16<<<2048, 256, 0, stream>>>(W_down, wd, D * F / 4);

  // 2. LN1
  ln_bf16<<<Mrows, 256, 0, stream>>>(x, ln1_g, ln1_b, hbf);

  // 3. QKV projection: [4096,6144] = h[4096,2048] * Wqkv^T
  gemm_wmma<0><<<dim3(TD / 128, Mrows / 128), 256, 0, stream>>>(
      hbf, wq, nullptr, nullptr, qkv, Mrows, TD, D);

  // 4. transpose V
  transpose_v<<<8192, 256, 0, stream>>>(qkv, vT);

  // 5. attention
  flash_attn<<<(Bsz * H * (L / 16)) / 4, 128, 0, stream>>>(qkv, vT, att);

  // 6. output projection + residual: x2 = x + att * Wo^T
  gemm_wmma<1><<<dim3(D / 128, Mrows / 128), 256, 0, stream>>>(
      att, wo, x, x2, nullptr, Mrows, D, D);

  // 7. LN2
  ln_bf16<<<Mrows, 256, 0, stream>>>(x2, ln2_g, ln2_b, hbf);

  // 8. fused gate/up + SiLU
  gemm_gateup<<<dim3(F / 128, Mrows / 128), 256, 0, stream>>>(
      hbf, wg, wu, gu, Mrows, F, D);

  // 9. down projection + residual -> output
  gemm_wmma<1><<<dim3(D / 128, Mrows / 128), 256, 0, stream>>>(
      gu, wd, x2, out, nullptr, Mrows, D, F);
}
